// VectorQuantizer_13511967113909
// MI455X (gfx1250) — matrix operand order repaired and checked by reading, not compiled, not run on hardware
//
#include <hip/hip_runtime.h>

// ---------------------------------------------------------------------------
// VQ nearest-codebook search for MI455X (gfx1250, wave32, WMMA + TDM).
//   z:        (8, 256, 64, 64) f32   -> rows N = 32768, dim 256
//   codebook: (8192, 256) f32        -> K = 8192 codes
// Outputs (concatenated in d_out): z_q [32768*256] f32, idx [32768] i32.
// Strategy: split-bf16 (hi/lo) compensated GEMM  S = Z @ C^T  via
// v_wmma_f32_16x16x32_bf16 with 3 independent accumulator chains
// (hi*hi, lo*hi, hi*lo), codebook tiles double-buffered into LDS by the
// Tensor Data Mover (tensor_load_to_lds, TENSORcnt), fused row argmin of
// (||e||^2 - 2*S), then exact f32 gather of winning rows.
// ---------------------------------------------------------------------------

typedef unsigned short u16;
typedef unsigned int   u32;
typedef unsigned long long u64;
typedef __attribute__((ext_vector_type(16))) __bf16 v16bf;
typedef __attribute__((ext_vector_type(8)))  float  v8f;
typedef __attribute__((ext_vector_type(4)))  unsigned int v4u;
typedef __attribute__((ext_vector_type(8)))  int v8i;
typedef __attribute__((ext_vector_type(4)))  int v4i;

union Frag {              // one 16x32-bf16 WMMA operand: 16 bf16 per lane
  v16bf v;
  uint4 q[2];             // two 16-byte halves
};

#define N_ROWS   32768
#define K_CODES  8192
#define DIM      256
#define TILE_K   32                    // codes staged per LDS tile
#define ROW_PAD  264                   // 256 bf16 + 8 pad -> 528B row stride
#define PLANE_U16 (TILE_K * ROW_PAD)   // 8448 u16 = 16896 B per hi/lo plane
#define ROWS_PER_BLOCK 128             // 8 waves * 16 rows

__device__ __forceinline__ u16 f32_to_bf16_rn(float x) {
  u32 u = __float_as_uint(x);
  return (u16)((u + 0x7fffu + ((u >> 16) & 1u)) >> 16);
}

// Split x = hi + lo with hi = truncate-to-bf16(x); lo rounded-nearest bf16.
__device__ __forceinline__ void split_bf16(float x, u16& hi, u16& lo) {
  u32 u = __float_as_uint(x);
  float fhi = __uint_as_float(u & 0xffff0000u);
  hi = (u16)(u >> 16);
  lo = f32_to_bf16_rn(x - fhi);
}

// --------------------------- preprocessing ---------------------------------

// z (b,c,h,w) -> row-major zHi/zLo [N][256] bf16. One block per row n.
__global__ void __launch_bounds__(256)
vq_prep_z(const float* __restrict__ z, u16* __restrict__ zHi,
          u16* __restrict__ zLo) {
  const int n = blockIdx.x;          // 0..32767  (= b*4096 + h*64 + w)
  const int c = threadIdx.x;         // 0..255
  const int b  = n >> 12;
  const int hw = n & 4095;
  float x = z[((size_t)(b * 256 + c) << 12) + hw];
  u16 hi, lo;
  split_bf16(x, hi, lo);
  zHi[((size_t)n << 8) + c] = hi;
  zLo[((size_t)n << 8) + c] = lo;
}

// codebook -> cHi/cLo [K][256] bf16 + eNorm[K] = ||e||^2 (f32, exact).
__global__ void __launch_bounds__(256)
vq_prep_cb(const float* __restrict__ cb, u16* __restrict__ cHi,
           u16* __restrict__ cLo, float* __restrict__ eNorm) {
  const int k = blockIdx.x;          // 0..8191
  const int c = threadIdx.x;         // 0..255
  float x = cb[((size_t)k << 8) + c];
  u16 hi, lo;
  split_bf16(x, hi, lo);
  cHi[((size_t)k << 8) + c] = hi;
  cLo[((size_t)k << 8) + c] = lo;

  __shared__ float red[256];
  red[c] = x * x;
  __syncthreads();
  #pragma unroll
  for (int s = 128; s > 0; s >>= 1) {
    if (c < s) red[c] += red[c + s];
    __syncthreads();
  }
  if (c == 0) eNorm[k] = red[0];
}

// ------------------------- TDM tile load (wave 0) --------------------------
// One 2D descriptor: 32 rows x 128 dwords (= 256 bf16) from global, padded
// in LDS by 4 dwords every 128 dwords -> 528B row stride (bank-conflict-free).
__device__ __forceinline__ void tdm_load_tile(const u16* gsrc, u16* ldsDst) {
  v4u g0;
  v8i g1;
  v4i gz4 = {0, 0, 0, 0};
  v8i gz8 = {0, 0, 0, 0, 0, 0, 0, 0};
  const u32 lds = (u32)(size_t)ldsDst;       // low 32 bits = LDS byte offset
  const u64 ga  = (u64)(size_t)gsrc;
  g0[0] = 1u;                                            // count=1 (valid D#)
  g0[1] = lds;                                           // lds_addr
  g0[2] = (u32)(ga & 0xffffffffu);                       // global_addr[31:0]
  g0[3] = (u32)((ga >> 32) & 0x1ffffffu) | (2u << 30);   // [56:32] | type=2
  g1[0] = (int)((2u << 16)     // data_size = 4B
              | (1u << 20)     // pad_enable
              | (6u << 22)     // pad_interval: 2^(6+1) = 128 dwords
              | (3u << 25));   // pad_amount: 4 dwords (16B)
  g1[1] = (int)(128u << 16);   // tensor_dim0 = 128 dwords (low16 in [31:16])
  g1[2] = (int)(32u  << 16);   // tensor_dim0 hi=0 | tensor_dim1 = 32 rows
  g1[3] = (int)(128u << 16);   // tensor_dim1 hi=0 | tile_dim0 = 128 dwords
  g1[4] = 32;                  // tile_dim1 = 32 rows | tile_dim2 = 0
  g1[5] = 128;                 // tensor_dim0_stride = 128 dwords (low 32)
  g1[6] = 0;                   // stride0 hi | tensor_dim1_stride low
  g1[7] = 0;
  // 6-arg variant (therock/clang-23 headers): groups 0..3 + extra group + cpol
  __builtin_amdgcn_tensor_load_to_lds(g0, g1, gz4, gz4, gz8, 0);
}

// ------------------------------ main kernel --------------------------------

__global__ void __launch_bounds__(256)
vq_argmin(const u16* __restrict__ zHi, const u16* __restrict__ zLo,
          const u16* __restrict__ cHi, const u16* __restrict__ cLo,
          const float* __restrict__ eNorm, const float* __restrict__ codebook,
          float* __restrict__ outQ, int* __restrict__ outIdx) {
  extern __shared__ u16 smem[];  // 2 buffers x (hi plane + lo plane) = 67584B

  const int tid  = threadIdx.x;
  const int wid  = tid >> 5;
  const int lane = tid & 31;
  const int l16  = lane & 15;
  const int hsel = lane >> 4;              // 0: low half, 1: high half

  const int rowBase = blockIdx.x * ROWS_PER_BLOCK + wid * 16;

  // ---- register-resident A fragments: 16 rows x 256 dims, hi + lo --------
  // A 16x32 bf16 layout: lane<16 holds row M=l16 K=[kb..kb+7, kb+16..kb+23];
  // lane>=16 holds same row K=[kb+8..kb+15, kb+24..kb+31].
  Frag aHi[8], aLo[8];
  {
    const size_t r = (size_t)(rowBase + l16) << 8;
    #pragma unroll
    for (int f = 0; f < 8; ++f) {
      const size_t base = r + f * 32 + hsel * 8;
      aHi[f].q[0] = *(const uint4*)(zHi + base);
      aHi[f].q[1] = *(const uint4*)(zHi + base + 16);
      aLo[f].q[0] = *(const uint4*)(zLo + base);
      aLo[f].q[1] = *(const uint4*)(zLo + base + 16);
    }
  }

  float best[8];
  int   bidx[8];
  #pragma unroll
  for (int j = 0; j < 8; ++j) { best[j] = 3.4e38f; bidx[j] = 0; }

  const int T = K_CODES / TILE_K;          // 256 tiles

  // Prologue: wave 0 DMAs tile 0 into buffer 0 (hi + lo planes).
  if (wid == 0) {
    tdm_load_tile(cHi, smem);
    tdm_load_tile(cLo, smem + PLANE_U16);
  }

  for (int t = 0; t < T; ++t) {
    const int cur = t & 1;
    u16* sHi = smem + cur * (2 * PLANE_U16);
    u16* sLo = sHi + PLANE_U16;

    if (wid == 0) {
      if (t + 1 < T) {
        // Prefetch tile t+1 into the other buffer (safe: all waves finished
        // reading it at the trailing barrier of iteration t-1).
        u16* nHi = smem + (cur ^ 1) * (2 * PLANE_U16);
        tdm_load_tile(cHi + (size_t)(t + 1) * TILE_K * DIM, nHi);
        tdm_load_tile(cLo + (size_t)(t + 1) * TILE_K * DIM, nHi + PLANE_U16);
        __builtin_amdgcn_s_wait_tensorcnt(2);  // tile t's 2 ops retired
      } else {
        __builtin_amdgcn_s_wait_tensorcnt(0);
      }
    }
    __syncthreads();                         // tile t visible to all waves

    const int tileBase = t * TILE_K;

    // ---- two 16-code sub-tiles, each 16x16x256 via 8x3 WMMAs --------------
    #pragma unroll
    for (int sub = 0; sub < 2; ++sub) {
      v8f acc0 = {}, acc1 = {}, acc2 = {};   // 3 independent WMMA chains
      // B 32x16 bf16 layout: lane's column = code row; lane<16 holds
      // K=[kb..kb+7, kb+16..kb+23], lane>=16 K=[kb+8..kb+15, kb+24..kb+31].
      const int crOff = (sub * 16 + l16) * ROW_PAD + hsel * 8;
      #pragma unroll
      for (int f = 0; f < 8; ++f) {
        const u16* ph = sHi + crOff + f * 32;
        const u16* pl = sLo + crOff + f * 32;
        Frag bh, bl;
        bh.q[0] = *(const uint4*)(ph);
        bh.q[1] = *(const uint4*)(ph + 16);
        bl.q[0] = *(const uint4*)(pl);
        bl.q[1] = *(const uint4*)(pl + 16);
        acc0 = __builtin_amdgcn_wmma_f32_16x16x32_bf16(
            false, aHi[f].v, false, bh.v, (short)0, acc0, false, false);
        acc1 = __builtin_amdgcn_wmma_f32_16x16x32_bf16(
            false, aLo[f].v, false, bh.v, (short)0, acc1, false, false);
        acc2 = __builtin_amdgcn_wmma_f32_16x16x32_bf16(
            false, aHi[f].v, false, bl.v, (short)0, acc2, false, false);
      }
      const int  code = tileBase + sub * 16 + l16;
      const float e   = eNorm[code];
      #pragma unroll
      for (int j = 0; j < 8; ++j) {
        // distance (minus constant ||z||^2): ||e||^2 - 2 * <z, e>
        float s = acc0[j] + acc1[j] + acc2[j];
        float d = __builtin_fmaf(-2.0f, s, e);
        bool c_  = d < best[j];
        bidx[j] = c_ ? code : bidx[j];
        best[j] = c_ ? d : best[j];
      }
    }
    __syncthreads();   // all waves done with buffer `cur` before its reuse
  }

  // ---- cross-lane min+argmin within each 16-lane half ----------------------
  // C/D layout: VGPR j: lanes 0-15 -> row rowBase+j, lanes 16-31 -> +j+8.
  #pragma unroll
  for (int j = 0; j < 8; ++j) {
    float b = best[j];
    int   i = bidx[j];
    #pragma unroll
    for (int m = 8; m > 0; m >>= 1) {
      float ob = __shfl_xor(b, m);
      int   oi = __shfl_xor(i, m);
      bool  c_ = (ob < b) || (ob == b && oi < i);
      b = c_ ? ob : b;
      i = c_ ? oi : i;
    }
    const int row = rowBase + j + (hsel << 3);
    if (l16 == 0) outIdx[row] = i;

    // broadcast winners of both halves; gather exact f32 codebook rows
    const int i0 = __shfl(i, 0);
    const int i1 = __shfl(i, 16);
    {
      const float4* src = (const float4*)(codebook + ((size_t)i0 << 8));
      float4* dst = (float4*)(outQ + ((size_t)(rowBase + j) << 8));
      dst[lane]      = src[lane];
      dst[lane + 32] = src[lane + 32];
    }
    {
      const float4* src = (const float4*)(codebook + ((size_t)i1 << 8));
      float4* dst = (float4*)(outQ + ((size_t)(rowBase + j + 8) << 8));
      dst[lane]      = src[lane];
      dst[lane + 32] = src[lane + 32];
    }
  }
}

// ------------------------------- launcher ----------------------------------

extern "C" void kernel_launch(void* const* d_in, const int* in_sizes, int n_in,
                              void* d_out, int out_size, void* d_ws, size_t ws_size,
                              hipStream_t stream) {
  const float* z  = (const float*)d_in[0];   // 8*256*64*64
  const float* cb = (const float*)d_in[1];   // 8192*256

  // workspace layout (bf16 planes + norms), ~40 MB
  u16* zHi = (u16*)d_ws;
  u16* zLo = zHi + (size_t)N_ROWS * DIM;
  u16* cHi = zLo + (size_t)N_ROWS * DIM;
  u16* cLo = cHi + (size_t)K_CODES * DIM;
  float* eN = (float*)(cLo + (size_t)K_CODES * DIM);

  float* outQ  = (float*)d_out;
  int*   outIx = (int*)(outQ + (size_t)N_ROWS * DIM);

  const size_t ldsBytes = 4 * PLANE_U16 * sizeof(u16);  // 67584 B

  vq_prep_z<<<N_ROWS, 256, 0, stream>>>(z, zHi, zLo);
  vq_prep_cb<<<K_CODES, 256, 0, stream>>>(cb, cHi, cLo, eN);
  vq_argmin<<<N_ROWS / ROWS_PER_BLOCK, 256, ldsBytes, stream>>>(
      zHi, zLo, cHi, cLo, eN, cb, outQ, outIx);
}
